// TransformerEncoderBlock_49065706390333
// MI455X (gfx1250) — matrix operand order repaired and checked by reading, not compiled, not run on hardware
//
#include <hip/hip_runtime.h>
#include <hip/hip_bf16.h>

// ---------------------------------------------------------------------------
// Transformer encoder block for MI455X (gfx1250, wave32, WMMA f16->f32).
// B=4 S=2048 D=768 H=8 HD=96 EXP=4.  GEMMs via v_wmma_f32_16x16x32_f16;
// all LDS tile staging via the Tensor Data Mover (tensor_load_to_lds),
// double-buffered with one barrier per K-step.
// ---------------------------------------------------------------------------

typedef __attribute__((ext_vector_type(16))) _Float16 v16h;
typedef __attribute__((ext_vector_type(8)))  _Float16 h8;     // 16 bytes
typedef __attribute__((ext_vector_type(8)))  float    v8f;
typedef __attribute__((ext_vector_type(4))) unsigned int u32x4;
typedef __attribute__((ext_vector_type(8))) int         i32x8;
typedef __attribute__((ext_vector_type(4))) int         i32x4;

union Frag16 { v16h v; h8 h[2]; };

static __device__ __forceinline__ v8f wmma_f16(v16h a, v16h b, v8f c) {
    return __builtin_amdgcn_wmma_f32_16x16x32_f16(false, a, false, b, (short)0, c,
                                                  false, false);
}

// TDM 2-D tile load: global (tile_y rows x tile_x elems, row stride
// stride_elems, 2-byte elements) -> contiguous LDS at lds_addr.
// D# layout per CDNA5 ISA ch.8 (group0 128b, group1 256b, trailing groups 0).
// This toolchain exposes the 6-arg builtin (extra zero-filled i32x8 group).
static __device__ __forceinline__ void tdm_load_2d(
    unsigned lds_addr, const void* gptr, int tile_x, int tile_y, int stride_elems) {
    unsigned long long ga = (unsigned long long)gptr;
    u32x4 g0;
    g0[0] = 1u;                                    // count=1, user mode
    g0[1] = lds_addr;                              // lds_addr [63:32]
    g0[2] = (unsigned)(ga & 0xffffffffull);        // global_addr[31:0]
    g0[3] = (unsigned)((ga >> 32) & 0x01ffffffull) // global_addr[56:32]
            | (2u << 30);                          // type=2 ("image")
    i32x8 g1;
    g1[0] = (1 << 16);                             // data_size=1 (2 bytes)
    g1[1] = (tile_x & 0xffff) << 16;               // tensor_dim0[15:0]
    g1[2] = ((unsigned)tile_x >> 16) | ((tile_y & 0xffff) << 16); // d0 hi | d1 lo
    g1[3] = ((unsigned)tile_y >> 16) | ((tile_x & 0xffff) << 16); // d1 hi | tile_dim0
    g1[4] = (tile_y & 0xffff);                     // tile_dim1 (tile_dim2=0)
    g1[5] = stride_elems;                          // tensor_dim0_stride[31:0]
    g1[6] = 0;                                     // stride0 hi | stride1 lo
    g1[7] = 0;
    i32x4 gz4 = {0, 0, 0, 0};
    i32x8 gz8 = {0, 0, 0, 0, 0, 0, 0, 0};
    __builtin_amdgcn_tensor_load_to_lds(g0, g1, gz4, gz4, gz8, 0);
}

static __device__ __forceinline__ unsigned lds_off(const void* p) {
    return (unsigned)(unsigned long long)p;        // flat->LDS drops upper bits
}

constexpr int Bc  = 4;
constexpr int Sc  = 2048;
constexpr int Dc  = 768;
constexpr int Hc  = 8;
constexpr int HDc = 96;
constexpr int MSc = Bc * Sc;     // 8192 rows
constexpr int DFc = Dc * 4;      // 3072

// ---------------------------------------------------------------------------
// f32 -> f16 convert (grid-stride)
// ---------------------------------------------------------------------------
__global__ void cvt_f32_f16(const float* __restrict__ in, _Float16* __restrict__ out,
                            int n) {
    int i = blockIdx.x * blockDim.x + threadIdx.x;
    for (; i < n; i += gridDim.x * blockDim.x) out[i] = (_Float16)in[i];
}

// ---------------------------------------------------------------------------
// f32 W[K][N] -> f16 Wt[N][K] transpose-convert (32x32 tiles via LDS).
// ---------------------------------------------------------------------------
__global__ __launch_bounds__(256) void cvt_transpose(
    const float* __restrict__ in, _Float16* __restrict__ out, int K, int N) {
    __shared__ float tile[32][33];
    const int k0 = blockIdx.y * 32, n0 = blockIdx.x * 32;
    const int x = threadIdx.x & 31, y = threadIdx.x >> 5;   // y: 0..7
    #pragma unroll
    for (int j = 0; j < 4; ++j)
        tile[y + 8 * j][x] = in[(size_t)(k0 + y + 8 * j) * N + n0 + x];
    __syncthreads();
    #pragma unroll
    for (int j = 0; j < 4; ++j)
        out[(size_t)(n0 + y + 8 * j) * K + k0 + x] = (_Float16)tile[x][y + 8 * j];
}

// ---------------------------------------------------------------------------
// Tiled WMMA GEMM: C[M,N] = A[M,K] (f16 row-major) @ Wt[N][K]^T (pre-transposed)
// Block tile 128x128, 8 waves of 32x64, K staged 32 wide.
// TDM stages both tiles; double-buffered; one barrier per K-step.
// EPI 0: Q/K -> head-major f16 [B,H,S,HD]
// EPI 1: bias + residual -> f32 (O-proj / FFN2)
// EPI 2: bias + GELU -> f16 (FFN1, stride DFc)
// EPI 3: V -> transposed f16 [B,H,HD,S]
// ---------------------------------------------------------------------------
template <int EPI>
__global__ __launch_bounds__(256) void gemm_f16(
    const _Float16* __restrict__ A, const _Float16* __restrict__ Wt,
    const float* __restrict__ bias, const float* __restrict__ resid,
    _Float16* __restrict__ outH, float* __restrict__ outF, int N, int K) {

    __shared__ alignas(32) _Float16 As[2][128][32];
    __shared__ alignas(32) _Float16 Bs[2][128][32];   // Bs[n][k]

    const int r0 = blockIdx.y * 128;
    const int n0 = blockIdx.x * 128;

    const int wave = threadIdx.x >> 5;
    const int lane = threadIdx.x & 31;
    const int half = lane >> 4;
    const int lm   = lane & 15;
    const int wm   = wave >> 1;   // 0..3  (32-row strip)
    const int wn   = wave & 1;    // 0..1  (64-col strip)

    auto stage = [&](int buf, int k0) {
        tdm_load_2d(lds_off(&As[buf][0][0]), A  + (size_t)r0 * K + k0, 32, 128, K);
        tdm_load_2d(lds_off(&Bs[buf][0][0]), Wt + (size_t)n0 * K + k0, 32, 128, K);
    };

    v8f acc[2][4];
    #pragma unroll
    for (int mi = 0; mi < 2; ++mi)
        #pragma unroll
        for (int ni = 0; ni < 4; ++ni) acc[mi][ni] = {};

    if (wave == 0) stage(0, 0);
    const int nsteps = K >> 5;
    for (int s = 0; s < nsteps; ++s) {
        const int buf = s & 1;
        if (wave == 0) __builtin_amdgcn_s_wait_tensorcnt(0);
        __syncthreads();                      // buffer `buf` now visible to all
        if (wave == 0 && s + 1 < nsteps) stage(buf ^ 1, (s + 1) * 32);

        Frag16 af[2]; v16h bf[4];
        #pragma unroll
        for (int mi = 0; mi < 2; ++mi) {
            int m = wm * 32 + mi * 16 + lm;
            af[mi].h[0] = *(const h8*)&As[buf][m][half * 16];
            af[mi].h[1] = *(const h8*)&As[buf][m][half * 16 + 8];
        }
        #pragma unroll
        for (int ni = 0; ni < 4; ++ni) {
            int n = wn * 64 + ni * 16 + lm;
            bf[ni] = *(const v16h*)&Bs[buf][n][half * 16];
        }
        #pragma unroll
        for (int mi = 0; mi < 2; ++mi)
            #pragma unroll
            for (int ni = 0; ni < 4; ++ni)
                acc[mi][ni] = wmma_f16(af[mi].v, bf[ni], acc[mi][ni]);
    }

    // ---- epilogue: C layout lane l -> (m = v + 8*(l/16), n = l%16) ----
    #pragma unroll
    for (int mi = 0; mi < 2; ++mi)
        #pragma unroll
        for (int ni = 0; ni < 4; ++ni)
            #pragma unroll
            for (int v = 0; v < 8; ++v) {
                int row = r0 + wm * 32 + mi * 16 + v + 8 * half;
                int col = n0 + wn * 64 + ni * 16 + lm;
                float val = acc[mi][ni][v] + bias[col];
                if constexpr (EPI == 0) {
                    int b = row >> 11, ss = row & (Sc - 1);
                    int head = col / HDc, hd = col % HDc;
                    outH[(((size_t)b * Hc + head) * Sc + ss) * HDc + hd] = (_Float16)val;
                } else if constexpr (EPI == 1) {
                    val += resid[(size_t)row * Dc + col];
                    outF[(size_t)row * Dc + col] = val;
                } else if constexpr (EPI == 2) {
                    float t = val;
                    float g = 0.5f * t *
                              (1.0f + tanhf(0.7978845608f * (t + 0.044715f * t * t * t)));
                    outH[(size_t)row * DFc + col] = (_Float16)g;
                } else {   // V transposed [B,H,HD,S]
                    int b = row >> 11, ss = row & (Sc - 1);
                    int head = col / HDc, hd = col % HDc;
                    outH[(((size_t)b * Hc + head) * HDc + hd) * Sc + ss] = (_Float16)val;
                }
            }
}

// ---------------------------------------------------------------------------
// Flash attention.  grid = B*H*16, block = 256 (8 waves x 16 query rows).
// Q/K head-major f16 [B,H,S,96]; V pre-transposed [B,H,96,S];
// output f16 [B,S,768].  K/V tiles staged by TDM, double-buffered.
// ---------------------------------------------------------------------------
__global__ __launch_bounds__(256) void attn_kernel(
    const _Float16* __restrict__ qh, const _Float16* __restrict__ kh,
    const _Float16* __restrict__ vt, _Float16* __restrict__ o) {

    __shared__ alignas(32) _Float16 Ks[2][64][96];    // keys row-major
    __shared__ alignas(32) _Float16 Vts[2][96][64];   // V transposed tile
    __shared__ alignas(16) _Float16 Ps[8][16][32];    // per-wave P tile

    const int bh   = blockIdx.x >> 4;
    const int qblk = blockIdx.x & 15;
    const int b = bh >> 3, h = bh & 7;

    const int wave = threadIdx.x >> 5;
    const int lane = threadIdx.x & 31;
    const int half = lane >> 4;
    const int lm   = lane & 15;
    const int q0   = qblk * 128 + wave * 16;

    const _Float16* qb  = qh + ((size_t)(b * Hc + h) * Sc) * HDc;
    const _Float16* kb  = kh + ((size_t)(b * Hc + h) * Sc) * HDc;
    const _Float16* vtb = vt + ((size_t)(b * Hc + h) * HDc) * Sc;

    auto stageKV = [&](int buf, int kc) {
        tdm_load_2d(lds_off(&Ks[buf][0][0]),  kb + (size_t)kc * HDc, 96, 64, HDc);
        tdm_load_2d(lds_off(&Vts[buf][0][0]), vtb + kc,              64, 96, Sc);
    };

    // Q fragments: 16 rows x 96 (3 chunks of K=32), registers for whole pass.
    Frag16 qf[3];
    #pragma unroll
    for (int c = 0; c < 3; ++c) {
        const _Float16* qp = qb + (size_t)(q0 + lm) * HDc + c * 32;
        qf[c].h[0] = *(const h8*)(qp + half * 16);
        qf[c].h[1] = *(const h8*)(qp + half * 16 + 8);
    }

    v8f Oacc[6];
    #pragma unroll
    for (int i = 0; i < 6; ++i) Oacc[i] = {};
    float mrow[8], lrow[8];
    #pragma unroll
    for (int v = 0; v < 8; ++v) { mrow[v] = -1e30f; lrow[v] = 0.0f; }

    const float scale = 0.1020620726f;   // 1/sqrt(96)

    if (wave == 0) stageKV(0, 0);
    for (int kc = 0; kc < Sc; kc += 64) {
        const int buf = (kc >> 6) & 1;
        if (wave == 0) __builtin_amdgcn_s_wait_tensorcnt(0);
        __syncthreads();
        if (wave == 0 && kc + 64 < Sc) stageKV(buf ^ 1, kc + 64);

        #pragma unroll
        for (int j2 = 0; j2 < 2; ++j2) {        // two 32-key groups
            // ---- S = Q K^T for 32 keys (two 16-col tiles) ----
            v8f s0 = {}, s1 = {};
            #pragma unroll
            for (int c = 0; c < 3; ++c) {
                v16h bk0 = *(const v16h*)&Ks[buf][j2 * 32 + lm][c * 32 + half * 16];
                s0 = wmma_f16(qf[c].v, bk0, s0);
                v16h bk1 = *(const v16h*)&Ks[buf][j2 * 32 + 16 + lm][c * 32 + half * 16];
                s1 = wmma_f16(qf[c].v, bk1, s1);
            }

            // ---- online softmax over the 32 keys ----
            float alpha[8];
            #pragma unroll
            for (int v = 0; v < 8; ++v) {
                float a0 = s0[v] * scale, a1 = s1[v] * scale;
                float mx = fmaxf(a0, a1);
                mx = fmaxf(mx, __shfl_xor(mx, 8));
                mx = fmaxf(mx, __shfl_xor(mx, 4));
                mx = fmaxf(mx, __shfl_xor(mx, 2));
                mx = fmaxf(mx, __shfl_xor(mx, 1));
                float mn = fmaxf(mrow[v], mx);
                float al = __expf(mrow[v] - mn);
                float p0 = __expf(a0 - mn), p1 = __expf(a1 - mn);
                float rs = p0 + p1;
                rs += __shfl_xor(rs, 8);
                rs += __shfl_xor(rs, 4);
                rs += __shfl_xor(rs, 2);
                rs += __shfl_xor(rs, 1);
                lrow[v] = lrow[v] * al + rs;
                mrow[v] = mn;
                alpha[v] = al;
                int m = v + 8 * half;
                Ps[wave][m][lm]      = (_Float16)p0;
                Ps[wave][m][16 + lm] = (_Float16)p1;
            }
            #pragma unroll
            for (int nc = 0; nc < 6; ++nc)
                #pragma unroll
                for (int v = 0; v < 8; ++v) Oacc[nc][v] *= alpha[v];

            // ---- re-layout P (C-layout -> A-fragment) via wave-local LDS ----
            Frag16 pf;
            pf.h[0] = *(const h8*)&Ps[wave][lm][half * 16];
            pf.h[1] = *(const h8*)&Ps[wave][lm][half * 16 + 8];

            // ---- O += P @ V  (6 tiles of 16 cols over HD=96) ----
            #pragma unroll
            for (int nc = 0; nc < 6; ++nc) {
                v16h bv = *(const v16h*)&Vts[buf][nc * 16 + lm][j2 * 32 + half * 16];
                Oacc[nc] = wmma_f16(pf.v, bv, Oacc[nc]);
            }
        }
    }

    // ---- write O / l  as f16 [B,S,768] with heads interleaved ----
    #pragma unroll
    for (int v = 0; v < 8; ++v) {
        float inv = 1.0f / lrow[v];
        int s = q0 + v + 8 * half;
        #pragma unroll
        for (int nc = 0; nc < 6; ++nc)
            o[((size_t)b * Sc + s) * Dc + h * HDc + nc * 16 + lm] =
                (_Float16)(Oacc[nc][v] * inv);
    }
}

// ---------------------------------------------------------------------------
// LayerNorm over rows of 768.  Optionally emits an f16 copy.
// ---------------------------------------------------------------------------
__global__ __launch_bounds__(256) void ln_kernel(
    const float* __restrict__ in, const float* __restrict__ g,
    const float* __restrict__ be, float* __restrict__ outF,
    _Float16* __restrict__ outH) {

    __shared__ float red[256];
    const int row = blockIdx.x, tid = threadIdx.x;
    const float* p = in + (size_t)row * Dc;
    float l0 = p[tid], l1 = p[tid + 256], l2 = p[tid + 512];
    red[tid] = l0 + l1 + l2;
    __syncthreads();
    for (int o2 = 128; o2 > 0; o2 >>= 1) {
        if (tid < o2) red[tid] += red[tid + o2];
        __syncthreads();
    }
    float mu = red[0] * (1.0f / Dc);
    __syncthreads();
    float d0 = l0 - mu, d1 = l1 - mu, d2 = l2 - mu;
    red[tid] = d0 * d0 + d1 * d1 + d2 * d2;
    __syncthreads();
    for (int o2 = 128; o2 > 0; o2 >>= 1) {
        if (tid < o2) red[tid] += red[tid + o2];
        __syncthreads();
    }
    float inv = rsqrtf(red[0] * (1.0f / Dc) + 1e-5f);

    float o0  = d0 * inv * g[tid]       + be[tid];
    float o1  = d1 * inv * g[tid + 256] + be[tid + 256];
    float o2v = d2 * inv * g[tid + 512] + be[tid + 512];
    float* po = outF + (size_t)row * Dc;
    po[tid] = o0; po[tid + 256] = o1; po[tid + 512] = o2v;
    if (outH) {
        _Float16* ph = outH + (size_t)row * Dc;
        ph[tid] = (_Float16)o0; ph[tid + 256] = (_Float16)o1;
        ph[tid + 512] = (_Float16)o2v;
    }
}

// ---------------------------------------------------------------------------
extern "C" void kernel_launch(void* const* d_in, const int* in_sizes, int n_in,
                              void* d_out, int out_size, void* d_ws, size_t ws_size,
                              hipStream_t stream) {
    (void)in_sizes; (void)n_in; (void)out_size; (void)ws_size;

    const float* x   = (const float*)d_in[0];
    const float* Wq  = (const float*)d_in[1];
    const float* bq  = (const float*)d_in[2];
    const float* Wk  = (const float*)d_in[3];
    const float* bk  = (const float*)d_in[4];
    const float* Wv  = (const float*)d_in[5];
    const float* bv  = (const float*)d_in[6];
    const float* Wo  = (const float*)d_in[7];
    const float* bo  = (const float*)d_in[8];
    const float* g1  = (const float*)d_in[9];
    const float* be1 = (const float*)d_in[10];
    const float* W1  = (const float*)d_in[11];
    const float* bf1 = (const float*)d_in[12];
    const float* W2  = (const float*)d_in[13];
    const float* bf2 = (const float*)d_in[14];
    const float* g2  = (const float*)d_in[15];
    const float* be2 = (const float*)d_in[16];
    float* out = (float*)d_out;

    size_t off = 0;
    auto alloc = [&](size_t bytes) -> void* {
        void* p = (void*)((char*)d_ws + off);
        off += (bytes + 255) & ~(size_t)255;
        return p;
    };
    _Float16* xh   = (_Float16*)alloc((size_t)MSc * Dc * 2);
    _Float16* Wqt  = (_Float16*)alloc((size_t)Dc * Dc * 2);     // [N][K]
    _Float16* Wkt  = (_Float16*)alloc((size_t)Dc * Dc * 2);
    _Float16* Wvt  = (_Float16*)alloc((size_t)Dc * Dc * 2);
    _Float16* Wot  = (_Float16*)alloc((size_t)Dc * Dc * 2);
    _Float16* W1t  = (_Float16*)alloc((size_t)Dc * DFc * 2);    // [3072][768]
    _Float16* W2t  = (_Float16*)alloc((size_t)DFc * Dc * 2);    // [768][3072]
    _Float16* qhm  = (_Float16*)alloc((size_t)MSc * Dc * 2);    // [B,H,S,96]
    _Float16* khm  = (_Float16*)alloc((size_t)MSc * Dc * 2);
    _Float16* vtm  = (_Float16*)alloc((size_t)MSc * Dc * 2);    // [B,H,96,S]
    _Float16* atth = (_Float16*)alloc((size_t)MSc * Dc * 2);    // [B,S,768]
    float*    tmp1 = (float*)   alloc((size_t)MSc * Dc * 4);
    float*    x1f  = (float*)   alloc((size_t)MSc * Dc * 4);
    _Float16* x1h  = (_Float16*)alloc((size_t)MSc * Dc * 2);
    _Float16* hb   = (_Float16*)alloc((size_t)MSc * DFc * 2);
    float*    tmp2 = (float*)   alloc((size_t)MSc * Dc * 4);

    dim3 blk(256);
    // 1) activation convert + weight transpose-converts (once per launch)
    cvt_f32_f16<<<1024, blk, 0, stream>>>(x, xh, MSc * Dc);
    cvt_transpose<<<dim3(Dc / 32, Dc / 32),  blk, 0, stream>>>(Wq, Wqt, Dc, Dc);
    cvt_transpose<<<dim3(Dc / 32, Dc / 32),  blk, 0, stream>>>(Wk, Wkt, Dc, Dc);
    cvt_transpose<<<dim3(Dc / 32, Dc / 32),  blk, 0, stream>>>(Wv, Wvt, Dc, Dc);
    cvt_transpose<<<dim3(Dc / 32, Dc / 32),  blk, 0, stream>>>(Wo, Wot, Dc, Dc);
    cvt_transpose<<<dim3(DFc / 32, Dc / 32), blk, 0, stream>>>(W1, W1t, Dc, DFc);
    cvt_transpose<<<dim3(Dc / 32, DFc / 32), blk, 0, stream>>>(W2, W2t, DFc, Dc);

    // 2) QKV projections (Q,K head-major; V transposed)
    gemm_f16<0><<<dim3(Dc / 128, MSc / 128), blk, 0, stream>>>(xh, Wqt, bq, nullptr, qhm, nullptr, Dc, Dc);
    gemm_f16<0><<<dim3(Dc / 128, MSc / 128), blk, 0, stream>>>(xh, Wkt, bk, nullptr, khm, nullptr, Dc, Dc);
    gemm_f16<3><<<dim3(Dc / 128, MSc / 128), blk, 0, stream>>>(xh, Wvt, bv, nullptr, vtm, nullptr, Dc, Dc);
    // 3) flash attention
    attn_kernel<<<Bc * Hc * 16, blk, 0, stream>>>(qhm, khm, vtm, atth);
    // 4) O-proj + residual(x)
    gemm_f16<1><<<dim3(Dc / 128, MSc / 128), blk, 0, stream>>>(atth, Wot, bo, x, nullptr, tmp1, Dc, Dc);
    // 5) LN1 -> x1 (f32 + f16)
    ln_kernel<<<MSc, blk, 0, stream>>>(tmp1, g1, be1, x1f, x1h);
    // 6) FFN1 + GELU -> f16
    gemm_f16<2><<<dim3(DFc / 128, MSc / 128), blk, 0, stream>>>(x1h, W1t, bf1, nullptr, hb, nullptr, DFc, Dc);
    // 7) FFN2 + residual(x1)
    gemm_f16<1><<<dim3(Dc / 128, MSc / 128), blk, 0, stream>>>(hb, W2t, bf2, x1f, nullptr, tmp2, Dc, DFc);
    // 8) LN2 -> output
    ln_kernel<<<MSc, blk, 0, stream>>>(tmp2, g2, be2, out, nullptr);
}
